// GRUActor_23596550324466
// MI455X (gfx1250) — matrix operand order repaired and checked by reading, not compiled, not run on hardware
//
#include <hip/hip_runtime.h>
#include <hip/hip_bf16.h>

typedef __attribute__((ext_vector_type(16))) _Float16 v16h;
typedef __attribute__((ext_vector_type(8)))  _Float16 v8h;
typedef __attribute__((ext_vector_type(8)))  float    v8f;
typedef __attribute__((ext_vector_type(4)))  float    v4f;

#define TRAJ 50

// ---- LDS byte offsets (dynamic shared memory, manually packed) ----
#define OFF_W1F   0        // 16384 halfs (W1 256x64 as 8 kc x 4 nt B-fragments)
#define OFF_W2F   32768    // 4096  halfs (W2 64x64  as 2 kc x 4 nt)
#define OFF_WHHF  40960    // 12288 halfs (Whh 64x192 as 2 kc x 12 nt)
#define OFF_WOUTF 65536    // 1024  halfs (Wout 64x2 zero-padded to 64x16, 2 kc)
#define OFF_H     67584    // 8 waves * 16x64 halfs (h / z scratch, row-major)
#define OFF_WP    83968    // 8 waves * 16 rows * 2 f32
#define OFF_B1    84992    // 64 f32
#define OFF_B2    85248    // 64 f32
#define OFF_CBIH  85504    // 192 f32: bih + bhh combined
#define OFF_WIH   86272    // 2*192 f32
#define OFF_BOUT  87808    // 2 f32
#define SMEM_BYTES 87840

__device__ __forceinline__ float fsigmoid(float x) {
    return 1.0f / (1.0f + __expf(-x));
}
__device__ __forceinline__ float ftanh(float x) {
    float e = __expf(-2.0f * fabsf(x));
    float t = (1.0f - e) / (1.0f + e);
    return x >= 0.0f ? t : -t;
}

// Load a 16x32 f16 A-fragment from a row-major [16][64]-half LDS tile.
// ISA layout: M = lane%16, K = (i<8 ? i : i+8) + 8*(lane>=16) + 32*kc
__device__ __forceinline__ v16h load_a_lds(const _Float16* hw, int m, int colbase) {
    const _Float16* p = hw + m * 64 + colbase;
    v8h h0 = *(const v8h*)(p);
    v8h h1 = *(const v8h*)(p + 16);
    v16h a;
#pragma unroll
    for (int i = 0; i < 8; ++i) { a[i] = h0[i]; a[8 + i] = h1[i]; }
    return a;
}

extern "C" __global__ __launch_bounds__(256)
void gru_actor_kernel(const float* __restrict__ x,
                      const float* __restrict__ W1,  const float* __restrict__ b1,
                      const float* __restrict__ W2,  const float* __restrict__ b2,
                      const float* __restrict__ Wih, const float* __restrict__ bih,
                      const float* __restrict__ Whh, const float* __restrict__ bhh,
                      const float* __restrict__ Wout,const float* __restrict__ bout,
                      float* __restrict__ out)
{
    extern __shared__ char smem[];
    _Float16* w1f   = (_Float16*)(smem + OFF_W1F);
    _Float16* w2f   = (_Float16*)(smem + OFF_W2F);
    _Float16* whhf  = (_Float16*)(smem + OFF_WHHF);
    _Float16* woutf = (_Float16*)(smem + OFF_WOUTF);
    _Float16* hsm   = (_Float16*)(smem + OFF_H);
    float*    wpsm  = (float*)(smem + OFF_WP);
    float*    b1s   = (float*)(smem + OFF_B1);
    float*    b2s   = (float*)(smem + OFF_B2);
    float*    cbihs = (float*)(smem + OFF_CBIH);
    float*    wihs  = (float*)(smem + OFF_WIH);
    float*    bouts = (float*)(smem + OFF_BOUT);

    const int tid = threadIdx.x;

    // ---- Stage weights into LDS, pre-swizzled to wave32 B-fragment order ----
    // B-frag element [lane][i]: K = 8*(lane>=16) + (i&7) + 16*(i>>3), N = lane%16
    for (int e = tid; e < 16384; e += 256) {              // W1: kc 0..7, nt 0..3
        int i = e & 15, ln = (e >> 4) & 31, nt = (e >> 9) & 3, kc = e >> 11;
        int K = kc * 32 + (ln >> 4) * 8 + (i & 7) + (i >> 3) * 16;
        int N = nt * 16 + (ln & 15);
        w1f[e] = (_Float16)W1[K * 64 + N];
    }
    for (int e = tid; e < 4096; e += 256) {               // W2: kc 0..1, nt 0..3
        int i = e & 15, ln = (e >> 4) & 31, nt = (e >> 9) & 3, kc = e >> 11;
        int K = kc * 32 + (ln >> 4) * 8 + (i & 7) + (i >> 3) * 16;
        int N = nt * 16 + (ln & 15);
        w2f[e] = (_Float16)W2[K * 64 + N];
    }
    for (int e = tid; e < 12288; e += 256) {              // Whh: kc 0..1, nt 0..11
        int i = e & 15, ln = (e >> 4) & 31;
        int g = e >> 9;                                   // kc*12 + nt
        int nt = g % 12, kc = g / 12;
        int K = kc * 32 + (ln >> 4) * 8 + (i & 7) + (i >> 3) * 16;
        int N = nt * 16 + (ln & 15);
        whhf[e] = (_Float16)Whh[K * 192 + N];
    }
    for (int e = tid; e < 1024; e += 256) {               // Wout zero-padded 64x16
        int i = e & 15, ln = (e >> 4) & 31, kc = e >> 9;
        int K = kc * 32 + (ln >> 4) * 8 + (i & 7) + (i >> 3) * 16;
        int N = ln & 15;
        woutf[e] = (N < 2) ? (_Float16)Wout[K * 2 + N] : (_Float16)0.0f;
    }
    for (int e = tid; e < 64;  e += 256) b1s[e]  = b1[e];
    for (int e = tid; e < 64;  e += 256) b2s[e]  = b2[e];
    for (int e = tid; e < 192; e += 256) cbihs[e] = bih[e] + bhh[e];
    for (int e = tid; e < 384; e += 256) wihs[e] = Wih[e];
    if (tid < 2) bouts[tid] = bout[tid];
    __syncthreads();

    const int lane  = tid & 31;
    const int wave  = tid >> 5;
    const int n16   = lane & 15;
    const int hi    = lane >> 4;      // 0 / 1
    const int loff8 = hi * 8;
    const int rowBase = blockIdx.x * 128 + wave * 16;
    const int myRow   = rowBase + n16;
    const float* xrow = x + (size_t)myRow * 256;
    _Float16* hw = hsm + wave * 1024;          // this wave's 16x64 half tile
    float*    wps = wpsm + wave * 32;          // this wave's 16x2 f32 wp

    // ---------------- Layer 1: z = relu(x @ W1 + b1), K=256, N=64 ----------------
    v8f acc1[4];
#pragma unroll
    for (int nt = 0; nt < 4; ++nt) acc1[nt] = {};     // SRC2 inline 0
#pragma unroll
    for (int kc = 0; kc < 8; ++kc) {
        v16h a;
        {
            const float* p = xrow + kc * 32 + loff8;
            v4f f0 = *(const v4f*)(p);
            v4f f1 = *(const v4f*)(p + 4);
            v4f f2 = *(const v4f*)(p + 16);
            v4f f3 = *(const v4f*)(p + 20);
#pragma unroll
            for (int i = 0; i < 4; ++i) {
                a[i]      = (_Float16)f0[i];
                a[4 + i]  = (_Float16)f1[i];
                a[8 + i]  = (_Float16)f2[i];
                a[12 + i] = (_Float16)f3[i];
            }
        }
#pragma unroll
        for (int nt = 0; nt < 4; ++nt) {
            v16h b = *(const v16h*)(w1f + ((kc * 4 + nt) * 32 + lane) * 16);
            acc1[nt] = __builtin_amdgcn_wmma_f32_16x16x32_f16(
                false, a, false, b, (short)0, acc1[nt], false, false);
        }
    }
    // bias + relu -> z1 tile in LDS (f16, row-major [16][64])
#pragma unroll
    for (int nt = 0; nt < 4; ++nt) {
        float b = b1s[nt * 16 + n16];
#pragma unroll
        for (int v = 0; v < 8; ++v) {
            float z = acc1[nt][v] + b;
            z = z > 0.0f ? z : 0.0f;
            hw[(v + hi * 8) * 64 + nt * 16 + n16] = (_Float16)z;
        }
    }

    // ---------------- Layer 2: h0 = relu(z1 @ W2 + b2), K=64, N=64 ----------------
    v16h a0 = load_a_lds(hw, n16, loff8);
    v16h a1 = load_a_lds(hw, n16, 32 + loff8);
    v8f hc[4];                                  // h state in C/D layout, f32
#pragma unroll
    for (int nt = 0; nt < 4; ++nt) {
        v8f acc = {};
        v16h bm = *(const v16h*)(w2f + ((0 * 4 + nt) * 32 + lane) * 16);
        acc = __builtin_amdgcn_wmma_f32_16x16x32_f16(false, a0, false, bm, (short)0, acc, false, false);
        bm = *(const v16h*)(w2f + ((1 * 4 + nt) * 32 + lane) * 16);
        acc = __builtin_amdgcn_wmma_f32_16x16x32_f16(false, a1, false, bm, (short)0, acc, false, false);
        float b = b2s[nt * 16 + n16];
#pragma unroll
        for (int v = 0; v < 8; ++v) {
            float z = acc[v] + b;
            z = z > 0.0f ? z : 0.0f;
            hc[nt][v] = z;
            hw[(v + hi * 8) * 64 + nt * 16 + n16] = (_Float16)z;
        }
    }

    // ---------------- GRU rollout: 50 steps ----------------
    float wp0c[8], wp1c[8];                     // wp cache for my 8 rows (gi math)
    float wpacc[8];                             // lanes with n16<2: running wp[row][n16]
#pragma unroll
    for (int v = 0; v < 8; ++v) { wp0c[v] = 0.0f; wp1c[v] = 0.0f; wpacc[v] = 0.0f; }
    a0 = load_a_lds(hw, n16, loff8);
    a1 = load_a_lds(hw, n16, 32 + loff8);

    for (int t = 0; t < TRAJ; ++t) {
#pragma unroll
        for (int nt = 0; nt < 4; ++nt) {
            const int j = nt * 16 + n16;        // column within H*3 gate space
            // gh = h @ Whh (SRC2 inline 0; bhh folded into cbihs)
            v8f ar = {}, au = {}, an = {};
            v16h bm;
            bm = *(const v16h*)(whhf + ((0 * 12 + nt) * 32 + lane) * 16);
            ar = __builtin_amdgcn_wmma_f32_16x16x32_f16(false, a0, false, bm, (short)0, ar, false, false);
            bm = *(const v16h*)(whhf + ((1 * 12 + nt) * 32 + lane) * 16);
            ar = __builtin_amdgcn_wmma_f32_16x16x32_f16(false, a1, false, bm, (short)0, ar, false, false);
            bm = *(const v16h*)(whhf + ((0 * 12 + nt + 4) * 32 + lane) * 16);
            au = __builtin_amdgcn_wmma_f32_16x16x32_f16(false, a0, false, bm, (short)0, au, false, false);
            bm = *(const v16h*)(whhf + ((1 * 12 + nt + 4) * 32 + lane) * 16);
            au = __builtin_amdgcn_wmma_f32_16x16x32_f16(false, a1, false, bm, (short)0, au, false, false);
            bm = *(const v16h*)(whhf + ((0 * 12 + nt + 8) * 32 + lane) * 16);
            an = __builtin_amdgcn_wmma_f32_16x16x32_f16(false, a0, false, bm, (short)0, an, false, false);
            bm = *(const v16h*)(whhf + ((1 * 12 + nt + 8) * 32 + lane) * 16);
            an = __builtin_amdgcn_wmma_f32_16x16x32_f16(false, a1, false, bm, (short)0, an, false, false);

            // gi = wp @ Wih + (bih + bhh), gates in C/D register layout
            const float wi0r = wihs[j],        wi1r = wihs[192 + j];
            const float wi0u = wihs[j + 64],   wi1u = wihs[256 + j];
            const float wi0n = wihs[j + 128],  wi1n = wihs[320 + j];
            const float cr = cbihs[j], cu = cbihs[j + 64], cn = cbihs[j + 128];
#pragma unroll
            for (int v = 0; v < 8; ++v) {
                float ir  = wp0c[v] * wi0r + wp1c[v] * wi1r + cr;
                float iz  = wp0c[v] * wi0u + wp1c[v] * wi1u + cu;
                float inn = wp0c[v] * wi0n + wp1c[v] * wi1n + cn;
                float r = fsigmoid(ir + ar[v]);
                float u = fsigmoid(iz + au[v]);
                float n = ftanh(inn + r * an[v]);
                float hnew = (1.0f - u) * n + u * hc[nt][v];
                hc[nt][v] = hnew;
                hw[(v + hi * 8) * 64 + j] = (_Float16)hnew;
            }
        }

        // A-fragments of h_new: used by the head below AND by the next step.
        a0 = load_a_lds(hw, n16, loff8);
        a1 = load_a_lds(hw, n16, 32 + loff8);

        // Head as WMMA: dx = sigmoid(h_new @ WoutPad + bout) in columns 0,1.
        v8f dacc = {};
        {
            v16h bm = *(const v16h*)(woutf + (lane) * 16);
            dacc = __builtin_amdgcn_wmma_f32_16x16x32_f16(false, a0, false, bm, (short)0, dacc, false, false);
            bm = *(const v16h*)(woutf + (32 + lane) * 16);
            dacc = __builtin_amdgcn_wmma_f32_16x16x32_f16(false, a1, false, bm, (short)0, dacc, false, false);
        }
        // Lanes with N-column 0/1 own dx for 8 rows each (no WMMA inside this branch).
        if (n16 < 2) {
            const float bo = bouts[n16];
#pragma unroll
            for (int v = 0; v < 8; ++v) {
                wpacc[v] += fsigmoid(dacc[v] + bo);
                const int row = v + hi * 8;
                out[(size_t)(rowBase + row) * 100 + t * 2 + n16] = wpacc[v];
                wps[2 * row + n16] = wpacc[v];
            }
        }

        // refresh per-lane wp cache for the next step's gi math
#pragma unroll
        for (int v = 0; v < 8; ++v) {
            wp0c[v] = wps[2 * (hi * 8 + v)];
            wp1c[v] = wps[2 * (hi * 8 + v) + 1];
        }
    }
}

extern "C" void kernel_launch(void* const* d_in, const int* in_sizes, int n_in,
                              void* d_out, int out_size, void* d_ws, size_t ws_size,
                              hipStream_t stream) {
    const float* x    = (const float*)d_in[0];
    const float* W1   = (const float*)d_in[1];
    const float* b1   = (const float*)d_in[2];
    const float* W2   = (const float*)d_in[3];
    const float* b2   = (const float*)d_in[4];
    const float* Wih  = (const float*)d_in[5];
    const float* bih  = (const float*)d_in[6];
    const float* Whh  = (const float*)d_in[7];
    const float* bhh  = (const float*)d_in[8];
    const float* Wout = (const float*)d_in[9];
    const float* bout = (const float*)d_in[10];
    float* out = (float*)d_out;

    const int rows = in_sizes[0] / 256;        // 32768
    dim3 grid(rows / 128);                     // 128 rows per block (8 waves x 16)
    dim3 block(256);
    gru_actor_kernel<<<grid, block, SMEM_BYTES, stream>>>(
        x, W1, b1, W2, b2, Wih, bih, Whh, bhh, Wout, bout, out);
}
